// DenoisingModule_44023414784602
// MI455X (gfx1250) — matrix operand order repaired and checked by reading, not compiled, not run on hardware
//
#include <hip/hip_runtime.h>
#include <hip/hip_bf16.h>
#include <math.h>

typedef __bf16 bf16;
typedef __bf16 v16bf __attribute__((ext_vector_type(16)));
typedef float  v8f   __attribute__((ext_vector_type(8)));

#define WMMA_BF16(a, b, c) \
  __builtin_amdgcn_wmma_f32_16x16x32_bf16(false, (a), false, (b), (short)0, (c), false, false)

constexpr int BN = 4, C = 256, HH = 64, WW = 64, HW = HH * WW;

// 32-byte LDS/global load with only 16B alignment guaranteed
static __device__ inline v16bf ldv16(const bf16* p) {
  v16bf r;
  __builtin_memcpy(&r, p, 32);
  return r;
}

// async copy of 16 bytes global -> LDS (no data VGPRs, tracked by ASYNCcnt)
static __device__ inline void async_cp16(unsigned ldsoff, const bf16* src) {
  asm volatile("global_load_async_to_lds_b128 %0, %1, off"
               :: "v"(ldsoff), "v"(src) : "memory");
}

// ---------------------------------------------------------------------------
// Kernel A: theta/phi projections (WMMA GEMM W[256x256] x X[256x4096]) and
// bf16 copy of values g = x. thetaT/phiT stored [HW, C].
// ---------------------------------------------------------------------------
__global__ __launch_bounds__(128)
void proj_kernel(const float* __restrict__ x,
                 const float* __restrict__ tw, const float* __restrict__ tb,
                 const float* __restrict__ pw, const float* __restrict__ pb,
                 bf16* __restrict__ thetaT, bf16* __restrict__ phiT,
                 bf16* __restrict__ g)
{
  const int lane  = threadIdx.x & 31;
  const int wave  = threadIdx.x >> 5;
  const int kh    = lane >> 4;
  const int nn    = lane & 15;
  const int nb    = blockIdx.x >> 8;
  const int pt    = blockIdx.x & 255;
  const int pbase = pt * 16;
  const int p     = pbase + nn;
  const float* xb = x + (size_t)nb * C * HW;

  v8f accT[4] = {};
  v8f accP[4] = {};

  for (int ks = 0; ks < 8; ++ks) {
    const int k0 = ks * 32;
    v16bf bv;
#pragma unroll
    for (int i = 0; i < 16; ++i)
      bv[i] = (bf16)xb[(size_t)(k0 + kh * 8 + i + (i >= 8 ? 8 : 0)) * HW + p];

#pragma unroll
    for (int mt = 0; mt < 4; ++mt) {
      const int Mrow = (mt * 4 + wave) * 16 + nn;
      const float* w0 = tw + (size_t)Mrow * C + k0 + kh * 8;
      v16bf av;
#pragma unroll
      for (int t = 0; t < 8; ++t) { av[t] = (bf16)w0[t]; av[8 + t] = (bf16)w0[16 + t]; }
      accT[mt] = WMMA_BF16(av, bv, accT[mt]);

      const float* w1 = pw + (size_t)Mrow * C + k0 + kh * 8;
      v16bf ap;
#pragma unroll
      for (int t = 0; t < 8; ++t) { ap[t] = (bf16)w1[t]; ap[8 + t] = (bf16)w1[16 + t]; }
      accP[mt] = WMMA_BF16(ap, bv, accP[mt]);
    }
  }

#pragma unroll
  for (int mt = 0; mt < 4; ++mt) {
    const int Mbase = (mt * 4 + wave) * 16;
#pragma unroll
    for (int r = 0; r < 8; ++r) {
      const int m = Mbase + r + 8 * kh;
      const size_t o = ((size_t)nb * HW + p) * C + m;
      thetaT[o] = (bf16)(accT[mt][r] + tb[m]);
      phiT[o]   = (bf16)(accP[mt][r] + pb[m]);
    }
  }

  for (int e = threadIdx.x; e < C * 16; e += 128) {
    const int c = e >> 4, pp = pbase + (e & 15);
    g[(size_t)nb * C * HW + (size_t)c * HW + pp] = (bf16)xb[(size_t)c * HW + pp];
  }
}

// ---------------------------------------------------------------------------
// Kernel B: fused flash attention, 8 waves / block, one 16-query tile per
// wave. 32-key steps; K/V tiles staged cooperatively into double-buffered LDS
// via global_load_async_to_lds_b128 (ASYNCcnt), shared by all 8 waves.
// S = Q K (16 WMMA/step from LDS), online softmax (wave32 shfl), P re-layout
// D->A through LDS (s_wait_dscnt), O += P V (16 WMMA/step from LDS).
// ---------------------------------------------------------------------------
__global__ __launch_bounds__(256)
void attn_kernel(const bf16* __restrict__ thetaT, const bf16* __restrict__ phiT,
                 const bf16* __restrict__ g, bf16* __restrict__ fT)
{
  // padded strides: phi row 264 el (132 dw), g row 40 el (20 dw) ->
  // conflict-free 16-lane b128 reads
  __shared__ __align__(32) bf16 phs[2][32 * 264];   // keys x channels
  __shared__ __align__(32) bf16 gvs[2][256 * 40];   // channels x keys
  __shared__ v16bf qsh[8][8][32];                   // Q in A-operand order
  __shared__ bf16  psh[8][16][32];                  // P transpose scratch

  const int tid   = threadIdx.x;
  const int lane  = tid & 31;
  const int wave  = tid >> 5;
  const int kh    = lane >> 4;
  const int nn    = lane & 15;
  const int qtile = blockIdx.x * 8 + wave;
  const int nb    = blockIdx.x >> 5;     // 32 blocks per batch
  const int qt    = qtile & 255;
  const int qrow0 = qt * 16;

  const bf16* thT = thetaT + (size_t)nb * HW * C;
  const bf16* phB = phiT   + (size_t)nb * HW * C;
  const bf16* gB  = g      + (size_t)nb * C * HW;

  // issue async staging of one 32-key step into LDS buffer `buf`
  auto stage = [&](int buf, int kt) {
    const int qb = kt * 32;
    const unsigned pbase = (unsigned)(uintptr_t)&phs[buf][0];
    const unsigned gbase = (unsigned)(uintptr_t)&gvs[buf][0];
#pragma unroll
    for (int i = 0; i < 4; ++i) {              // phi rows: 32 x 256 bf16
      const int chunk = tid + i * 256;
      const int q = chunk >> 5, ch = chunk & 31;
      async_cp16(pbase + (unsigned)(q * 264 + ch * 8) * 2,
                 phB + (size_t)(qb + q) * C + ch * 8);
    }
#pragma unroll
    for (int i = 0; i < 4; ++i) {              // g rows: 256 x 32 bf16
      const int chunk = tid + i * 256;
      const int c = chunk >> 2, ch = chunk & 3;
      async_cp16(gbase + (unsigned)(c * 40 + ch * 8) * 2,
                 gB + (size_t)c * HW + qb + ch * 8);
    }
  };

  // stage Q (16 x 256) into LDS in WMMA A layout per k-step
  {
    const bf16* qrow = thT + (size_t)(qrow0 + nn) * C;
#pragma unroll
    for (int ks = 0; ks < 8; ++ks) {
      const bf16* q0 = qrow + ks * 32 + kh * 16;
      v16bf av;
#pragma unroll
      for (int t = 0; t < 8; ++t) { av[t] = q0[t]; av[8 + t] = q0[8 + t]; }
      qsh[wave][ks][lane] = av;
    }
  }
  asm volatile("s_wait_dscnt 0x0" ::: "memory");

  stage(0, 0);   // prologue: first key tile in flight

  v8f O[16] = {};
  float mrow[8], lsum[8];
#pragma unroll
  for (int r = 0; r < 8; ++r) { mrow[r] = -__builtin_inff(); lsum[r] = 0.f; }

  for (int kt = 0; kt < 128; ++kt) {
    const int cur = kt & 1;
    const int qb  = kt * 32; (void)qb;

    if (kt + 1 < 128) {
      stage(cur ^ 1, kt + 1);                         // prefetch next tile
      asm volatile("s_wait_asynccnt 0x8" ::: "memory"); // current 8 done (in-order)
    } else {
      asm volatile("s_wait_asynccnt 0x0" ::: "memory");
    }
    __syncthreads();   // current buffer visible to all waves

    const bf16* ph = &phs[cur][0];
    const bf16* gv = &gvs[cur][0];

    v8f sA = {}, sB = {};
#pragma unroll
    for (int ks = 0; ks < 8; ++ks) {
      v16bf a  = qsh[wave][ks][lane];
      v16bf b0 = ldv16(ph + (nn)      * 264 + ks * 32 + kh * 16);
      v16bf b1 = ldv16(ph + (16 + nn) * 264 + ks * 32 + kh * 16);
      sA = WMMA_BF16(a, b0, sA);
      sB = WMMA_BF16(a, b1, sB);
    }

    const float scale = 0.0625f;   // 1/sqrt(256)
    float rm[8], pA[8], pB[8], rs[8], corr[8];
#pragma unroll
    for (int r = 0; r < 8; ++r) {
      const float a = sA[r] * scale, b = sB[r] * scale;
      sA[r] = a; sB[r] = b;
      rm[r] = fmaxf(a, b);
    }
#pragma unroll
    for (int mk = 1; mk <= 8; mk <<= 1)
#pragma unroll
      for (int r = 0; r < 8; ++r)
        rm[r] = fmaxf(rm[r], __shfl_xor(rm[r], mk, 32));

#pragma unroll
    for (int r = 0; r < 8; ++r) {
      const float mn = fmaxf(mrow[r], rm[r]);
      corr[r] = __expf(mrow[r] - mn);
      mrow[r] = mn;
      pA[r] = __expf(sA[r] - mn);
      pB[r] = __expf(sB[r] - mn);
      rs[r] = pA[r] + pB[r];
    }
#pragma unroll
    for (int mk = 1; mk <= 8; mk <<= 1)
#pragma unroll
      for (int r = 0; r < 8; ++r)
        rs[r] += __shfl_xor(rs[r], mk, 32);
#pragma unroll
    for (int r = 0; r < 8; ++r) lsum[r] = lsum[r] * corr[r] + rs[r];

#pragma unroll
    for (int ct = 0; ct < 16; ++ct)
#pragma unroll
      for (int r = 0; r < 8; ++r) O[ct][r] *= corr[r];

    // P: D layout -> LDS -> A layout (wave-private region)
#pragma unroll
    for (int r = 0; r < 8; ++r) {
      const int pq = r + 8 * kh;
      psh[wave][pq][nn]      = (bf16)pA[r];
      psh[wave][pq][16 + nn] = (bf16)pB[r];
    }
    asm volatile("s_wait_dscnt 0x0" ::: "memory");
    v16bf ap;
#pragma unroll
    for (int t = 0; t < 8; ++t) {
      ap[t]     = psh[wave][nn][kh * 16 + t];
      ap[8 + t] = psh[wave][nn][kh * 16 + 8 + t];
    }
    asm volatile("" ::: "memory");

    // O[p, c] += P[p, q] * V[q, c]   (B from staged g tile)
#pragma unroll
    for (int ct = 0; ct < 16; ++ct) {
      v16bf bv = ldv16(gv + (ct * 16 + nn) * 40 + kh * 16);
      O[ct] = WMMA_BF16(ap, bv, O[ct]);
    }

    __syncthreads();   // all waves done with buffer before it is re-filled
  }

  float inv[8];
#pragma unroll
  for (int r = 0; r < 8; ++r) inv[r] = 1.0f / lsum[r];
#pragma unroll
  for (int ct = 0; ct < 16; ++ct) {
    const int c = ct * 16 + nn;
#pragma unroll
    for (int r = 0; r < 8; ++r) {
      const int pq = qrow0 + r + 8 * kh;
      fT[((size_t)nb * HW + pq) * C + c] = (bf16)(O[ct][r] * inv[r]);
    }
  }
}

// ---------------------------------------------------------------------------
// Kernel C: result = x + conv1(f) (WMMA GEMM + bias + residual).
// ---------------------------------------------------------------------------
__global__ __launch_bounds__(128)
void out_kernel(const float* __restrict__ x, const float* __restrict__ cw,
                const float* __restrict__ cb, const bf16* __restrict__ fT,
                float* __restrict__ out)
{
  const int lane  = threadIdx.x & 31;
  const int wave  = threadIdx.x >> 5;
  const int kh    = lane >> 4;
  const int nn    = lane & 15;
  const int nb    = blockIdx.x >> 8;
  const int pt    = blockIdx.x & 255;
  const int pbase = pt * 16;
  const int p     = pbase + nn;

  v8f acc[4] = {};
  const bf16* frow = fT + ((size_t)nb * HW + p) * C;

  for (int ks = 0; ks < 8; ++ks) {
    const int k0 = ks * 32;
    v16bf bv = ldv16(frow + k0 + kh * 16);
#pragma unroll
    for (int mt = 0; mt < 4; ++mt) {
      const int Mrow = (mt * 4 + wave) * 16 + nn;
      const float* w0 = cw + (size_t)Mrow * C + k0 + kh * 16;
      v16bf av;
#pragma unroll
      for (int t = 0; t < 8; ++t) { av[t] = (bf16)w0[t]; av[8 + t] = (bf16)w0[8 + t]; }
      acc[mt] = WMMA_BF16(av, bv, acc[mt]);
    }
  }

#pragma unroll
  for (int mt = 0; mt < 4; ++mt) {
    const int Mbase = (mt * 4 + wave) * 16;
#pragma unroll
    for (int r = 0; r < 8; ++r) {
      const int m = Mbase + r + 8 * kh;
      const size_t idx = ((size_t)nb * C + m) * HW + p;
      out[idx] = x[idx] + cb[m] + acc[mt][r];
    }
  }
}

// ---------------------------------------------------------------------------
extern "C" void kernel_launch(void* const* d_in, const int* in_sizes, int n_in,
                              void* d_out, int out_size, void* d_ws, size_t ws_size,
                              hipStream_t stream) {
  const float* x  = (const float*)d_in[0];
  const float* tw = (const float*)d_in[1];
  const float* tb = (const float*)d_in[2];
  const float* pw = (const float*)d_in[3];
  const float* pb = (const float*)d_in[4];
  const float* cw = (const float*)d_in[5];
  const float* cb = (const float*)d_in[6];
  float* out = (float*)d_out;

  // workspace: thetaT | phiT | g | fT, each N*HW*C bf16 (8 MB) -> 32 MB total
  const size_t planeEl = (size_t)BN * HW * C;
  bf16* thetaT = (bf16*)d_ws;
  bf16* phiT   = thetaT + planeEl;
  bf16* g      = phiT + planeEl;
  bf16* fT     = g + planeEl;

  proj_kernel<<<dim3(BN * 256), dim3(128), 0, stream>>>(x, tw, tb, pw, pb,
                                                        thetaT, phiT, g);
  attn_kernel<<<dim3(BN * 32), dim3(256), 0, stream>>>(thetaT, phiT, g, fT);
  out_kernel<<<dim3(BN * 256), dim3(128), 0, stream>>>(x, cw, cb, fT, out);
}
